// AMGCR_22849226015114
// MI455X (gfx1250) — matrix operand order repaired and checked by reading, not compiled, not run on hardware
//
#include <hip/hip_runtime.h>
#include <hip/hip_bf16.h>

// ---------------- problem constants (from reference) ----------------
#define C_NU 100000
#define C_NI 50000
#define C_E  800000
#define C_D  64
#define C_B  1024
#define P_DROP 0.25f
#define INV_KEEP (1.0f/0.75f)
#define INV_TEMP 5.0f
#define L1C 0.2f
#define L2C 0.01f
#define L3C 1e-5f

typedef __attribute__((ext_vector_type(16))) _Float16 v16h;
typedef __attribute__((ext_vector_type(8)))  _Float16 v8h;
typedef __attribute__((ext_vector_type(4)))  _Float16 v4h;
typedef __attribute__((ext_vector_type(8)))  float    v8f;

__device__ __forceinline__ float sigmoidf_(float x){ return 1.0f/(1.0f+__expf(-x)); }
__device__ __forceinline__ void atomAddF(float* p, float v){
  __hip_atomic_fetch_add(p, v, __ATOMIC_RELAXED, __HIP_MEMORY_SCOPE_AGENT);
}

// ---------------- WMMA fragment builders (CDNA5 layouts, wide loads) ----------------
// A: 16x32 f16 tile from row-major [16][64] f16 (LDS or global). Per lane the 16
// halves are two contiguous 8-half runs -> two 16B loads + shuffle concat.
__device__ __forceinline__ v16h frag_a(const _Float16* tile, int lane, int ks){
  const int m  = lane & 15;
  const int kb = ks + ((lane & 16) ? 16 : 0);
  v8h lo = *(const v8h*)(tile + m*64 + kb);
  v8h hi = *(const v8h*)(tile + m*64 + kb + 8);
  return __builtin_shufflevector(lo, hi, 0,1,2,3,4,5,6,7,8,9,10,11,12,13,14,15);
}
// B from pre-permuted weight tile sWp[((nt*2+ks32)*32+lane)*16 + h] -> contiguous.
__device__ __forceinline__ v16h frag_b_wp(const _Float16* sWp, int lane, int ks32, int nt){
  const _Float16* p = sWp + (size_t)(((nt*2 + ks32)*32 + lane) << 4);
  v8h lo = *(const v8h*)(p);
  v8h hi = *(const v8h*)(p + 8);
  return __builtin_shufflevector(lo, hi, 0,1,2,3,4,5,6,7,8,9,10,11,12,13,14,15);
}
// B[k][n] = Emb_h[col_n][k] : contiguous 16 halves per lane in f16 global rows.
__device__ __forceinline__ v16h frag_b_embT(const _Float16* Emb, int row, int lane, int ks){
  const int kb = ks + ((lane & 16) ? 16 : 0);
  const _Float16* p = Emb + (size_t)row*64 + kb;
  v8h lo = *(const v8h*)(p);
  v8h hi = *(const v8h*)(p + 8);
  return __builtin_shufflevector(lo, hi, 0,1,2,3,4,5,6,7,8,9,10,11,12,13,14,15);
}
__device__ __forceinline__ v8f wmma_f16(v16h a, v16h b, v8f c){
  return __builtin_amdgcn_wmma_f32_16x16x32_f16(false, a, false, b, (short)0, c, false, false);
}
__device__ __forceinline__ v4h cvt4(float4 x){
  v4h r; r[0]=(_Float16)x.x; r[1]=(_Float16)x.y; r[2]=(_Float16)x.z; r[3]=(_Float16)x.w;
  return r;
}

// ---------------- utility kernels ----------------
__global__ __launch_bounds__(256) void k_zero(float* p, long n){
  long i = (long)blockIdx.x*256 + threadIdx.x;
  if (i < n) p[i] = 0.0f;
}
__global__ __launch_bounds__(256) void k_copy(float* dst, const float* src, long n){
  long i = (long)blockIdx.x*256 + threadIdx.x;
  if (i < n) dst[i] = src[i];
}
__global__ __launch_bounds__(256) void k_add(float* dst, const float* src, long n){
  long i = (long)blockIdx.x*256 + threadIdx.x;
  if (i < n) dst[i] += src[i];
}
// f32 -> f16 conversion (4 elements / thread), optional scale
__global__ __launch_bounds__(256) void k_cvt_h(const float* __restrict__ X,
                                               _Float16* __restrict__ Y, long n4, float scale){
  long t = (long)blockIdx.x*256 + threadIdx.x;
  if (t < n4){
    float4 x = ((const float4*)X)[t];
    x.x *= scale; x.y *= scale; x.z *= scale; x.w *= scale;
    ((v4h*)Y)[t] = cvt4(x);
  }
}
// gather 64-wide rows by id into f16 (scaled)
__global__ __launch_bounds__(256) void k_gather_h(const float* __restrict__ X,
    const int* __restrict__ ids, _Float16* __restrict__ out, int nB, float scale){
  long t = (long)blockIdx.x*256 + threadIdx.x;
  if (t < (long)nB*16){
    int b = (int)(t >> 4), c = (int)(t & 15);
    float4 x = *(const float4*)(X + (size_t)ids[b]*64 + c*4);
    x.x *= scale; x.y *= scale; x.z *= scale; x.w *= scale;
    ((v4h*)out)[t] = cvt4(x);
  }
}

// ---------------- SpMM: out[seg[e]] += w_e * X[gat[e]]  (D=64, float4 per thread) ----
__global__ __launch_bounds__(256) void k_spmm(const float* __restrict__ vals,
                                              const float* __restrict__ drop,
                                              const int* __restrict__ seg,
                                              const int* __restrict__ gat,
                                              const float* __restrict__ X,
                                              float* __restrict__ out, int nE){
  long t = (long)blockIdx.x*256 + threadIdx.x;
  long e = t >> 4;
  if (e >= nE) return;
  float w = vals[e];
  if (drop) w = (drop[e] > P_DROP) ? (w * INV_KEEP) : 0.0f;
  if (w == 0.0f) return;
  int c = (int)(t & 15);
  int s = seg[e], g = gat[e];
  float4 x = *(const float4*)(X + (size_t)g*64 + c*4);
  float* o = out + (size_t)s*64 + c*4;
  atomAddF(o+0, w*x.x); atomAddF(o+1, w*x.y);
  atomAddF(o+2, w*x.z); atomAddF(o+3, w*x.w);
}

// stage a 64x64 f32 weight into LDS pre-permuted into B-fragment order
__device__ __forceinline__ void stage_wp(const float* __restrict__ W, _Float16* sWp, int tid){
  for (int idx = tid; idx < 4096; idx += 256){
    int h    = idx & 15;
    int lane = (idx >> 4) & 31;
    int ks32 = (idx >> 9) & 1;
    int nt   = idx >> 10;
    int n = nt*16 + (lane & 15);
    int k = ks32*32 + ((lane & 16) ? 16 : 0) + h;
    sWp[idx] = (_Float16)W[k*64 + n];
  }
}

// ---------------- MLP + ATT views (WMMA over 16-edge tiles) ----------------
__global__ __launch_bounds__(256) void k_mlp_att(const float* __restrict__ Eu0,
    const float* __restrict__ Ei0, const int* __restrict__ src, const int* __restrict__ dst,
    const float* __restrict__ W1u, const float* __restrict__ W1i,
    const float* __restrict__ b1, const float* __restrict__ W2, const float* __restrict__ b2,
    const float* __restrict__ att_a, float* __restrict__ view_mlp,
    float* __restrict__ view_att, int nE){
  __shared__ _Float16 sWu[4096];
  __shared__ _Float16 sWi[4096];
  __shared__ float sb1[64], sW2[64], sa[128];
  __shared__ _Float16 sAu[8][1024];
  __shared__ _Float16 sAi[8][1024];
  const int tid = threadIdx.x;
  stage_wp(W1u, sWu, tid);
  stage_wp(W1i, sWi, tid);
  if (tid < 64){ sb1[tid] = b1[tid]; sW2[tid] = W2[tid]; }
  if (tid < 128) sa[tid] = att_a[tid];
  const int wave = tid >> 5, lane = tid & 31;
  const long base = (long)blockIdx.x*128 + wave*16;
  // stage 16 gathered rows (x2) as f16, float4 -> ds_store_b64
  for (int t = lane; t < 256; t += 32){
    int r = t >> 4, c = t & 15;
    long e = base + r;
    float4 xu = make_float4(0.f,0.f,0.f,0.f), xi = xu;
    if (e < nE){
      xu = *(const float4*)(Eu0 + (size_t)src[e]*64 + c*4);
      xi = *(const float4*)(Ei0 + (size_t)dst[e]*64 + c*4);
    }
    *(v4h*)&sAu[wave][r*64 + c*4] = cvt4(xu);
    *(v4h*)&sAi[wave][r*64 + c*4] = cvt4(xi);
  }
  __syncthreads();
  float p[8] = {0,0,0,0,0,0,0,0};
  const float b2v = b2[0];
  #pragma unroll
  for (int nt = 0; nt < 4; ++nt){
    v8f acc = {};
    #pragma unroll
    for (int ks32 = 0; ks32 < 2; ++ks32){
      acc = wmma_f16(frag_a(&sAu[wave][0], lane, ks32*32), frag_b_wp(sWu, lane, ks32, nt), acc);
      acc = wmma_f16(frag_a(&sAi[wave][0], lane, ks32*32), frag_b_wp(sWi, lane, ks32, nt), acc);
    }
    const int n = nt*16 + (lane & 15);
    const float w2n = sW2[n], b1n = sb1[n];
    #pragma unroll
    for (int r = 0; r < 8; ++r){
      float hv = acc[r] + b1n;
      hv = (hv > 0.0f) ? hv : 0.0f;
      p[r] += hv * w2n;
    }
  }
  #pragma unroll
  for (int r = 0; r < 8; ++r){
    float v = p[r];
    #pragma unroll
    for (int m = 1; m < 16; m <<= 1) v += __shfl_xor(v, m, 32);
    p[r] = v;
  }
  if ((lane & 15) == 0){
    const int mb = (lane & 16) ? 8 : 0;
    #pragma unroll
    for (int r = 0; r < 8; ++r){
      long e = base + mb + r;
      if (e < nE) view_mlp[e] = sigmoidf_(p[r] + b2v);
    }
  }
  if (lane < 16){
    long e = base + lane;
    if (e < nE){
      float s = 0.0f;
      const _Float16* au = &sAu[wave][lane*64];
      const _Float16* ai = &sAi[wave][lane*64];
      #pragma unroll
      for (int c = 0; c < 8; ++c){
        v8h xu = *(const v8h*)(au + c*8);
        v8h xi = *(const v8h*)(ai + c*8);
        #pragma unroll
        for (int j = 0; j < 8; ++j)
          s += (float)xu[j]*sa[c*8 + j] + (float)xi[j]*sa[64 + c*8 + j];
      }
      float lr = (s > 0.0f) ? s : 0.2f*s;
      view_att[e] = sigmoidf_(lr);
    }
  }
}

__global__ __launch_bounds__(256) void k_wv(const float* __restrict__ wv,
                                            float* __restrict__ out, int nE){
  long e = (long)blockIdx.x*256 + threadIdx.x;
  if (e < nE) out[e] = sigmoidf_(wv[e]);
}

// ---------------- Y = X @ W  ([N,64]@[64,64], WMMA) ----------------
__global__ __launch_bounds__(256) void k_rowgemm(const float* __restrict__ X,
    const float* __restrict__ W, float* __restrict__ Y, int Nrows){
  __shared__ _Float16 sWp[4096];
  __shared__ _Float16 sX[8][1024];
  const int tid = threadIdx.x;
  stage_wp(W, sWp, tid);
  const int wave = tid >> 5, lane = tid & 31;
  const long rb = (long)blockIdx.x*128 + wave*16;
  for (int t = lane; t < 256; t += 32){
    int r = t >> 4, c = t & 15;
    long row = rb + r;
    float4 x = make_float4(0.f,0.f,0.f,0.f);
    if (row < Nrows) x = *(const float4*)(X + row*64 + c*4);
    *(v4h*)&sX[wave][r*64 + c*4] = cvt4(x);
  }
  __syncthreads();
  #pragma unroll
  for (int nt = 0; nt < 4; ++nt){
    v8f acc = {};
    #pragma unroll
    for (int ks32 = 0; ks32 < 2; ++ks32)
      acc = wmma_f16(frag_a(&sX[wave][0], lane, ks32*32), frag_b_wp(sWp, lane, ks32, nt), acc);
    const int n  = nt*16 + (lane & 15);
    const int mb = (lane & 16) ? 8 : 0;
    #pragma unroll
    for (int r = 0; r < 8; ++r){
      long row = rb + mb + r;
      if (row < Nrows) Y[row*64 + n] = acc[r];
    }
  }
}

// ---------------- out[e] = sigmoid(dot(U[src[e]], V[dst[e]])) ----------------
__global__ __launch_bounds__(256) void k_edge_dot(const float* __restrict__ U,
    const float* __restrict__ V, const int* __restrict__ src, const int* __restrict__ dst,
    float* __restrict__ out, int nE){
  long e = (long)blockIdx.x*256 + threadIdx.x;
  if (e >= nE) return;
  const float4* a = (const float4*)(U + (size_t)src[e]*64);
  const float4* b = (const float4*)(V + (size_t)dst[e]*64);
  float s = 0.0f;
  #pragma unroll
  for (int i = 0; i < 16; ++i){
    float4 x = a[i], y = b[i];
    s += x.x*y.x + x.y*y.y + x.z*y.z + x.w*y.w;
  }
  out[e] = sigmoidf_(s);
}

// ---------------- softmax-over-edges denominators (4 views) ----------------
__global__ __launch_bounds__(256) void k_vsums(const float* __restrict__ views,
    const float* __restrict__ fw, const float* __restrict__ fb,
    float* __restrict__ sums, int nE){
  float loc[4] = {0,0,0,0};
  for (long e = (long)blockIdx.x*256 + threadIdx.x; e < nE; e += (long)gridDim.x*256){
    float w = fw[e], b = fb[e];
    #pragma unroll
    for (int v = 0; v < 4; ++v)
      loc[v] += __expf(tanhf(w*views[(size_t)v*nE + e] + b));
  }
  __shared__ float red[256];
  for (int v = 0; v < 4; ++v){
    red[threadIdx.x] = loc[v]; __syncthreads();
    for (int s = 128; s > 0; s >>= 1){
      if (threadIdx.x < s) red[threadIdx.x] += red[threadIdx.x + s];
      __syncthreads();
    }
    if (threadIdx.x == 0) atomAddF(&sums[v], red[0]);
    __syncthreads();
  }
}

// ---------------- fuse views -> Ag, baew, aug_vals, pr-loss partial ----------------
__global__ __launch_bounds__(256) void k_fuse(const float* __restrict__ views,
    const float* __restrict__ fw, const float* __restrict__ fb,
    const float* __restrict__ sums, const float* __restrict__ adj,
    float* __restrict__ baew /*in: pre*/, float* __restrict__ augv,
    float* __restrict__ acc_pr, int nE){
  long e = (long)blockIdx.x*256 + threadIdx.x;
  float locpr = 0.0f;
  if (e < nE){
    float w = fw[e], b = fb[e];
    float vsum = 0.0f, agw = 0.0f;
    #pragma unroll
    for (int v = 0; v < 4; ++v){
      float vv = views[(size_t)v*nE + e];
      float sw = __expf(tanhf(w*vv + b)) / sums[v];
      agw  += sw*vv;
      vsum += vv;
    }
    float Ag  = (vsum - 3.0f*agw) * 0.2f;   // (sum(views) - 3*Ag)/5
    float bae = baew[e] * Ag;                // pre * Ag
    baew[e] = bae;
    augv[e] = bae * adj[e];
    locpr   = -__logf(bae);
  }
  __shared__ float red[256];
  red[threadIdx.x] = locpr; __syncthreads();
  for (int s = 128; s > 0; s >>= 1){
    if (threadIdx.x < s) red[threadIdx.x] += red[threadIdx.x + s];
    __syncthreads();
  }
  if (threadIdx.x == 0) atomAddF(acc_pr, red[0]);
}

// ---- contrastive negatives: rowsum[b] += sum_cols exp((G/T)·Emb); all-f16, LDS-free ----
__global__ __launch_bounds__(256) void k_cl_neg(const _Float16* __restrict__ Gh,
    const _Float16* __restrict__ Eh, float* __restrict__ rowsum, int Ncols){
  const int tid  = threadIdx.x;
  const int wave = tid >> 5, lane = tid & 31;
  const int rowbase = blockIdx.x * 16;
  const int colbase = blockIdx.y*128 + wave*16;
  const int col = colbase + (lane & 15);
  const int colOK = (col < Ncols);
  const int rowe = colOK ? col : 0;
  const _Float16* Atile = Gh + (size_t)rowbase*64;
  v8f acc = {};
  #pragma unroll
  for (int ks = 0; ks < 64; ks += 32)
    acc = wmma_f16(frag_a(Atile, lane, ks), frag_b_embT(Eh, rowe, lane, ks), acc);
  float es[8];
  #pragma unroll
  for (int r = 0; r < 8; ++r) es[r] = colOK ? __expf(acc[r]) : 0.0f;
  #pragma unroll
  for (int r = 0; r < 8; ++r){
    float v = es[r];
    #pragma unroll
    for (int m = 1; m < 16; m <<= 1) v += __shfl_xor(v, m, 32);
    es[r] = v;
  }
  if ((lane & 15) == 0){
    const int mb = (lane & 16) ? 8 : 0;
    #pragma unroll
    for (int r = 0; r < 8; ++r) atomAddF(&rowsum[rowbase + mb + r], es[r]);
  }
}

// ---------------- positive CL terms + BPR (per sample) ----------------
__device__ __forceinline__ float dot64(const float* a, const float* b){
  const float4* x = (const float4*)a; const float4* y = (const float4*)b;
  float s = 0.0f;
  #pragma unroll
  for (int i = 0; i < 16; ++i){
    float4 u = x[i], v = y[i];
    s += u.x*v.x + u.y*v.y + u.z*v.z + u.w*v.w;
  }
  return s;
}
__global__ __launch_bounds__(256) void k_pos_bpr(const float* __restrict__ Zu,
    const float* __restrict__ Zi, const float* __restrict__ Su, const float* __restrict__ Si,
    const int* __restrict__ uids, const int* __restrict__ iids,
    const int* __restrict__ pos, const int* __restrict__ neg,
    float* __restrict__ scal, int nB){
  long b = (long)blockIdx.x*256 + threadIdx.x;
  float posv = 0.0f, bprv = 0.0f;
  if (b < nB){
    int u = uids[b], it = iids[b], pp = pos[b], nn = neg[b];
    float du = dot64(Zu + (size_t)u*64, Su + (size_t)u*64) * INV_TEMP;
    float di = dot64(Zi + (size_t)it*64, Si + (size_t)it*64) * INV_TEMP;
    du = fminf(5.0f, fmaxf(-5.0f, du));
    di = fminf(5.0f, fmaxf(-5.0f, di));
    posv = du + di;
    float diff = dot64(Su + (size_t)u*64, Si + (size_t)pp*64)
               - dot64(Su + (size_t)u*64, Si + (size_t)nn*64);
    bprv = (diff > 0.0f) ? __logf(1.0f + __expf(-diff))
                         : (-diff + __logf(1.0f + __expf(diff)));
  }
  __shared__ float r0[256], r1[256];
  r0[threadIdx.x] = posv; r1[threadIdx.x] = bprv; __syncthreads();
  for (int s = 128; s > 0; s >>= 1){
    if (threadIdx.x < s){ r0[threadIdx.x] += r0[threadIdx.x + s]; r1[threadIdx.x] += r1[threadIdx.x + s]; }
    __syncthreads();
  }
  if (threadIdx.x == 0){ atomAddF(&scal[6], r0[0]); atomAddF(&scal[4], r1[0]); }
}

// ---------------- sum of squares (L2 reg) ----------------
__global__ __launch_bounds__(256) void k_sumsq(const float* __restrict__ p, long n,
                                               float* __restrict__ acc){
  float s = 0.0f;
  for (long i = (long)blockIdx.x*256 + threadIdx.x; i < n; i += (long)gridDim.x*256){
    float v = p[i]; s += v*v;
  }
  __shared__ float red[256];
  red[threadIdx.x] = s; __syncthreads();
  for (int st = 128; st > 0; st >>= 1){
    if (threadIdx.x < st) red[threadIdx.x] += red[threadIdx.x + st];
    __syncthreads();
  }
  if (threadIdx.x == 0) atomAddF(acc, red[0]);
}

// ---------------- final loss assembly ----------------
__global__ __launch_bounds__(256) void k_final(const float* __restrict__ rsu,
    const float* __restrict__ rsi, const float* __restrict__ scal, float* __restrict__ out){
  float lsum = 0.0f;
  for (int b = threadIdx.x; b < C_B; b += 256)
    lsum += __logf(rsu[b] + 1e-8f) + __logf(rsi[b] + 1e-8f);
  __shared__ float red[256];
  red[threadIdx.x] = lsum; __syncthreads();
  for (int s = 128; s > 0; s >>= 1){
    if (threadIdx.x < s) red[threadIdx.x] += red[threadIdx.x + s];
    __syncthreads();
  }
  if (threadIdx.x == 0){
    float neg_sc   = red[0] / (float)C_B;
    float pos_sc   = scal[6] / (float)C_B;
    float loss_cl  = -pos_sc + neg_sc;
    float loss_bpr = scal[4] / (float)C_B;
    float loss_pr  = L2C * scal[5] / (float)C_E;
    float loss_reg = L3C * scal[7];
    float l1cl = L1C * loss_cl;
    out[0] = loss_bpr + l1cl + loss_pr + loss_reg;
    out[1] = loss_bpr;
    out[2] = l1cl;
    out[3] = loss_pr;
  }
}

// =========================== host-side sequencing ===========================
extern "C" void kernel_launch(void* const* d_in, const int* in_sizes, int n_in,
                              void* d_out, int out_size, void* d_ws, size_t ws_size,
                              hipStream_t stream) {
  const float* Eu0   = (const float*)d_in[0];
  const float* Ei0   = (const float*)d_in[1];
  const float* fusew = (const float*)d_in[2];
  const float* fuseb = (const float*)d_in[3];
  const float* W1u   = (const float*)d_in[4];
  const float* W1i   = (const float*)d_in[5];
  const float* b1    = (const float*)d_in[6];
  const float* W2    = (const float*)d_in[7];
  const float* b2    = (const float*)d_in[8];
  const float* atta  = (const float*)d_in[9];
  const float* wvp   = (const float*)d_in[10];
  const float* Wg    = (const float*)d_in[11];
  const float* adj   = (const float*)d_in[12];
  const float* dmain = (const float*)d_in[13];
  const float* daug  = (const float*)d_in[14];
  const int*   src   = (const int*)d_in[15];
  const int*   dst   = (const int*)d_in[16];
  const int*   uids  = (const int*)d_in[17];
  const int*   iids  = (const int*)d_in[18];
  const int*   pids  = (const int*)d_in[19];
  const int*   nids  = (const int*)d_in[20];
  float* out = (float*)d_out;

  const size_t NUD = (size_t)C_NU * C_D;   // 6,400,000
  const size_t NID = (size_t)C_NI * C_D;   // 3,200,000
  const size_t EE  = (size_t)C_E;

  float* ws = (float*)d_ws;
  float* Au = ws;            float* Bu = Au + NUD;
  float* Su = Bu + NUD;      float* Zu = Su + NUD;
  float* Ai = Zu + NUD;      float* Bi = Ai + NID;
  float* Si = Bi + NID;      float* Zi = Si + NID;
  float* views = Zi + NID;               // 4*E
  float* baew  = views + 4*EE;           // E
  float* augv  = baew + EE;              // E
  float* rsu   = augv + EE;              // B
  float* rsi   = rsu + C_B;              // B
  float* scal  = rsi + C_B;              // 16 scalars
  _Float16* Suh = (_Float16*)(scal + 16);         // NU*D halves
  _Float16* Sih = Suh + NUD;                      // NI*D halves
  _Float16* Ghu = Sih + NID;                      // B*D halves (pre-scaled by 1/T)
  _Float16* Ghi = Ghu + (size_t)C_B*C_D;

  const int gNU  = (int)((NUD + 255) / 256);
  const int gNI  = (int)((NID + 255) / 256);
  const int gE   = (int)((EE  + 255) / 256);
  const int gSp  = (int)((EE*16 + 255) / 256);
  const int gMLP = (int)((EE + 127) / 128);
  const int gRGu = (C_NU + 127) / 128;
  const dim3 blk(256);

  // ---- main propagate: S = E0 + layer1 + layer2 ----
  k_copy<<<gNU, blk, 0, stream>>>(Su, Eu0, (long)NUD);
  k_copy<<<gNI, blk, 0, stream>>>(Si, Ei0, (long)NID);
  k_zero<<<gNU, blk, 0, stream>>>(Au, (long)NUD);
  k_spmm<<<gSp, blk, 0, stream>>>(adj, dmain + 0*EE, src, dst, Ei0, Au, C_E);
  k_zero<<<gNI, blk, 0, stream>>>(Ai, (long)NID);
  k_spmm<<<gSp, blk, 0, stream>>>(adj, dmain + 1*EE, dst, src, Eu0, Ai, C_E);
  k_add<<<gNU, blk, 0, stream>>>(Su, Au, (long)NUD);
  k_add<<<gNI, blk, 0, stream>>>(Si, Ai, (long)NID);
  k_zero<<<gNU, blk, 0, stream>>>(Bu, (long)NUD);
  k_spmm<<<gSp, blk, 0, stream>>>(adj, dmain + 2*EE, src, dst, Ai, Bu, C_E);
  k_zero<<<gNI, blk, 0, stream>>>(Bi, (long)NID);
  k_spmm<<<gSp, blk, 0, stream>>>(adj, dmain + 3*EE, dst, src, Au, Bi, C_E);
  k_add<<<gNU, blk, 0, stream>>>(Su, Bu, (long)NUD);
  k_add<<<gNI, blk, 0, stream>>>(Si, Bi, (long)NID);

  // ---- views ----
  k_mlp_att<<<gMLP, blk, 0, stream>>>(Eu0, Ei0, src, dst, W1u, W1i, b1, W2, b2, atta,
                                      views + 0*EE, views + 3*EE, C_E);
  k_wv<<<gE, blk, 0, stream>>>(wvp, views + 1*EE, C_E);
  k_zero<<<gNU, blk, 0, stream>>>(Au, (long)NUD);
  k_spmm<<<gSp, blk, 0, stream>>>(adj, nullptr, src, dst, Si, Au, C_E);   // h_u
  k_zero<<<gNI, blk, 0, stream>>>(Ai, (long)NID);
  k_spmm<<<gSp, blk, 0, stream>>>(adj, nullptr, dst, src, Su, Ai, C_E);   // h_i
  k_rowgemm<<<gRGu, blk, 0, stream>>>(Au, Wg, Bu, C_NU);                  // HW = h_u @ Wg
  k_edge_dot<<<gE, blk, 0, stream>>>(Bu, Ai, src, dst, views + 2*EE, C_E);// gcn view
  k_edge_dot<<<gE, blk, 0, stream>>>(Su, Si, src, dst, baew, C_E);        // pre

  // ---- fuse (softmax over E) ----
  k_zero<<<9, blk, 0, stream>>>(rsu, (long)(C_B + C_B + 16));             // rsu,rsi,scal
  k_vsums<<<1024, blk, 0, stream>>>(views, fusew, fuseb, scal, C_E);
  k_fuse<<<gE, blk, 0, stream>>>(views, fusew, fuseb, scal, adj, baew, augv, &scal[5], C_E);

  // ---- aug propagate -> Z ----
  k_copy<<<gNU, blk, 0, stream>>>(Zu, Eu0, (long)NUD);
  k_copy<<<gNI, blk, 0, stream>>>(Zi, Ei0, (long)NID);
  k_zero<<<gNU, blk, 0, stream>>>(Au, (long)NUD);
  k_spmm<<<gSp, blk, 0, stream>>>(augv, daug + 0*EE, src, dst, Ei0, Au, C_E);
  k_zero<<<gNI, blk, 0, stream>>>(Ai, (long)NID);
  k_spmm<<<gSp, blk, 0, stream>>>(augv, daug + 1*EE, dst, src, Eu0, Ai, C_E);
  k_add<<<gNU, blk, 0, stream>>>(Zu, Au, (long)NUD);
  k_add<<<gNI, blk, 0, stream>>>(Zi, Ai, (long)NID);
  k_zero<<<gNU, blk, 0, stream>>>(Bu, (long)NUD);
  k_spmm<<<gSp, blk, 0, stream>>>(augv, daug + 2*EE, src, dst, Ai, Bu, C_E);
  k_zero<<<gNI, blk, 0, stream>>>(Bi, (long)NID);
  k_spmm<<<gSp, blk, 0, stream>>>(augv, daug + 3*EE, dst, src, Au, Bi, C_E);
  k_add<<<gNU, blk, 0, stream>>>(Zu, Bu, (long)NUD);
  k_add<<<gNI, blk, 0, stream>>>(Zi, Bi, (long)NID);

  // ---- contrastive loss ----
  k_cvt_h<<<(int)((NUD/4 + 255)/256), blk, 0, stream>>>(Su, Suh, (long)(NUD/4), 1.0f);
  k_cvt_h<<<(int)((NID/4 + 255)/256), blk, 0, stream>>>(Si, Sih, (long)(NID/4), 1.0f);
  k_gather_h<<<(C_B*16 + 255)/256, blk, 0, stream>>>(Zu, uids, Ghu, C_B, INV_TEMP);
  k_gather_h<<<(C_B*16 + 255)/256, blk, 0, stream>>>(Zi, iids, Ghi, C_B, INV_TEMP);
  k_cl_neg<<<dim3(C_B/16, (C_NU + 127)/128), blk, 0, stream>>>(Ghu, Suh, rsu, C_NU);
  k_cl_neg<<<dim3(C_B/16, (C_NI + 127)/128), blk, 0, stream>>>(Ghi, Sih, rsi, C_NI);
  k_pos_bpr<<<(C_B + 255)/256, blk, 0, stream>>>(Zu, Zi, Su, Si, uids, iids, pids, nids,
                                                 scal, C_B);

  // ---- L2 regularization over the 12 parameter tensors ----
  for (int pi = 0; pi < 12; ++pi){
    long n = in_sizes[pi];
    int g = (int)((n + 255) / 256);
    if (g > 2048) g = 2048;
    k_sumsq<<<g, blk, 0, stream>>>((const float*)d_in[pi], n, &scal[7]);
  }

  // ---- assemble losses ----
  k_final<<<1, blk, 0, stream>>>(rsu, rsi, scal, out);
  (void)n_in; (void)out_size; (void)ws_size;
}
